// MixtralAttention_15899968930374
// MI455X (gfx1250) — matrix operand order repaired and checked by reading, not compiled, not run on hardware
//
#include <hip/hip_runtime.h>

// ---------------------------------------------------------------------------
// MI455X (gfx1250) Mixtral attention layer: f16 WMMA everywhere.
//   hidden(f32) -> f16 ; weights(f32) -> f16 transposed (one-time)
//   [WMMA GEMM, async LDS staging] -> qkv(f16) -> RoPE/split ->
//   [WMMA flash attention] -> attn(f16) -> [WMMA GEMM] -> out(f32)
// ---------------------------------------------------------------------------

typedef __attribute__((ext_vector_type(16))) _Float16 v16h;
typedef __attribute__((ext_vector_type(8)))  float    v8f;
typedef __attribute__((ext_vector_type(4)))  _Float16 v4h;

union Frag {
    v16h     v;
    _Float16 h[16];
    int4     q[2];
};

__device__ __forceinline__ v8f vzero8() {
    v8f z = {0.f, 0.f, 0.f, 0.f, 0.f, 0.f, 0.f, 0.f};
    return z;
}

__device__ __forceinline__ v8f wmma_f16(v16h a, v16h b, v8f c) {
    // D = A(16x32 f16) * B(32x16 f16) + C(16x16 f32)
    return __builtin_amdgcn_wmma_f32_16x16x32_f16(
        /*neg_a=*/false, a, /*neg_b=*/false, b,
        /*c_mod=*/(short)0, c, /*reuse_a=*/false, /*reuse_b=*/false);
}

// Issue 64 contiguous bytes as 4 async b128 LDS<-global moves.
// ISA 15.18.3 op 98: the instruction offset is added to both the LDS and the global address.
__device__ __forceinline__ void async_copy64(void* lds, const void* g) {
    unsigned l = (unsigned)(uintptr_t)lds;          // flat addr[31:0] == LDS offset
    unsigned long long a = (unsigned long long)(uintptr_t)g;
    asm volatile(
        "global_load_async_to_lds_b128 %0, %1, off\n\t"
        "global_load_async_to_lds_b128 %0, %1, off offset:16\n\t"
        "global_load_async_to_lds_b128 %0, %1, off offset:32\n\t"
        "global_load_async_to_lds_b128 %0, %1, off offset:48"
        :: "v"(l), "v"(a) : "memory");
}
__device__ __forceinline__ void wait_async0() {
    asm volatile("s_wait_asynccnt 0x0" ::: "memory");
}

// ---------------------------------------------------------------------------
// f32 -> f16 conversion, vectorized x4 (activation path, no transpose)
// ---------------------------------------------------------------------------
__global__ __launch_bounds__(256) void cvt_f32_f16_x4(
    const float4* __restrict__ in, v4h* __restrict__ out, long n4) {
    long i = (long)blockIdx.x * blockDim.x + threadIdx.x;
    long stride = (long)gridDim.x * blockDim.x;
    for (; i < n4; i += stride) {
        float4 f = in[i];
        v4h h;
        h[0] = (_Float16)f.x; h[1] = (_Float16)f.y;
        h[2] = (_Float16)f.z; h[3] = (_Float16)f.w;
        out[i] = h;
    }
}

// ---------------------------------------------------------------------------
// One-time weight convert + transpose: W (K x N, f32) -> WT (N x K, f16).
// 64x64 tile via LDS (padded stride 65 to spread banks); coalesced f32 reads,
// coalesced b128 f16 writes. Moves the transpose cost out of the GEMM loop
// (which re-reads weights 32x).
// ---------------------------------------------------------------------------
__global__ __launch_bounds__(256) void cvt_transpose_f16(
    const float* __restrict__ W, _Float16* __restrict__ WT, int K, int N) {
    __shared__ _Float16 tile[64 * 65];          // [k][n]
    const int k0 = blockIdx.y * 64, n0 = blockIdx.x * 64;
    const int r = threadIdx.x >> 2;             // 0..63
    const int c4 = (threadIdx.x & 3) * 16;      // 0,16,32,48
    const float4* src = (const float4*)(W + (size_t)(k0 + r) * N + n0 + c4);
#pragma unroll
    for (int j = 0; j < 4; ++j) {
        float4 f = src[j];
        int c = c4 + j * 4;
        tile[r * 65 + c + 0] = (_Float16)f.x;
        tile[r * 65 + c + 1] = (_Float16)f.y;
        tile[r * 65 + c + 2] = (_Float16)f.z;
        tile[r * 65 + c + 3] = (_Float16)f.w;
    }
    __syncthreads();
    // write WT row n = n0+r, k-range k0+c4 .. +15 (two b128 stores)
    _Float16* dst = WT + (size_t)(n0 + r) * K + k0 + c4;
    union { int4 q; _Float16 h[8]; } o;
#pragma unroll
    for (int j = 0; j < 2; ++j) {
#pragma unroll
        for (int e = 0; e < 8; ++e) o.h[e] = tile[(c4 + j * 8 + e) * 65 + r];
        ((int4*)dst)[j] = o.q;
    }
}

// ---------------------------------------------------------------------------
// f16 GEMM: C(MxN) = A(MxK row-major) * BT(NxK row-major)^T, f32 accumulate.
// Block tile 128x128x64, 256 threads = 8 waves (2x4), wave tile 64x32 (4x2
// WMMA tiles). Both operands staged straight (no scatter) into LDS with
// async b128 LDS<-global moves; fragments are two ds_load_b128 each.
// ---------------------------------------------------------------------------
template <bool F16OUT, bool ASYNC>
__global__ __launch_bounds__(256) void gemm_f16_wmma(
    const _Float16* __restrict__ A, const _Float16* __restrict__ BT,
    void* __restrict__ Cout, int M, int N, int K) {
    __shared__ _Float16 Als[128 * 64];   // [m][k]
    __shared__ _Float16 Bls[128 * 64];   // [n][k]
    (void)M;

    const int tid  = threadIdx.x;
    const int lane = tid & 31;
    const int wave = tid >> 5;
    const int wm = wave >> 2, wn = wave & 3;
    const int bm0 = blockIdx.y * 128, bn0 = blockIdx.x * 128;
    const int m0 = wm * 64, n0 = wn * 32;

    const int lm  = lane & 15;
    const int hi  = lane >> 4;
    const int kbA = hi * 8;    // A frag: lanes 0-15 K0-7/16-23, 16-31 K8-15/24-31
    const int ksB = hi * 8;    // B frag: lanes 0-15 K0-7/16-23, 16-31 K8-15/24-31

    // staging: thread -> 64B of one row of each operand tile
    const int sr = tid >> 1;              // 0..127
    const int sc = (tid & 1) * 32;        // halves
    const _Float16* aSrc = A  + (size_t)(bm0 + sr) * K + sc;
    const _Float16* bSrc = BT + (size_t)(bn0 + sr) * K + sc;
    _Float16* aDst = Als + sr * 64 + sc;
    _Float16* bDst = Bls + sr * 64 + sc;

    v8f acc[4][2];
#pragma unroll
    for (int i = 0; i < 4; ++i)
#pragma unroll
        for (int j = 0; j < 2; ++j) acc[i][j] = vzero8();

    for (int k0 = 0; k0 < K; k0 += 64) {
        __syncthreads();                       // prior tile fully consumed
        if constexpr (ASYNC) {
            async_copy64(aDst, aSrc + k0);
            async_copy64(bDst, bSrc + k0);
            // prefetch next k-block into caches while this one lands
            int kp = (k0 + 64 < K) ? k0 + 64 : 0;
            __builtin_prefetch(aSrc + kp, 0, 3);
            __builtin_prefetch(bSrc + kp, 0, 3);
            wait_async0();
        } else {
            const int4* as = (const int4*)(aSrc + k0);
            const int4* bs = (const int4*)(bSrc + k0);
            int4* ad = (int4*)aDst;
            int4* bd = (int4*)bDst;
#pragma unroll
            for (int j = 0; j < 4; ++j) { ad[j] = as[j]; bd[j] = bs[j]; }
        }
        __syncthreads();

#pragma unroll
        for (int kk = 0; kk < 64; kk += 32) {
            Frag af[4], bf[2];
#pragma unroll
            for (int i = 0; i < 4; ++i) {
                const _Float16* p = Als + (m0 + i * 16 + lm) * 64 + kk;
                af[i].q[0] = *(const int4*)(p + kbA);
                af[i].q[1] = *(const int4*)(p + 16 + kbA);
            }
#pragma unroll
            for (int j = 0; j < 2; ++j) {
                const _Float16* p = Bls + (n0 + j * 16 + lm) * 64 + kk + ksB;
                bf[j].q[0] = ((const int4*)p)[0];
                bf[j].q[1] = ((const int4*)p)[2];
            }
#pragma unroll
            for (int i = 0; i < 4; ++i)
#pragma unroll
                for (int j = 0; j < 2; ++j)
                    acc[i][j] = wmma_f16(af[i].v, bf[j].v, acc[i][j]);
        }
    }

    // C layout: VGPR r -> row (r + 8*hi), col = lm within each 16x16 tile
#pragma unroll
    for (int i = 0; i < 4; ++i)
#pragma unroll
        for (int j = 0; j < 2; ++j)
#pragma unroll
            for (int r = 0; r < 8; ++r) {
                int row = bm0 + m0 + i * 16 + r + 8 * hi;
                int col = bn0 + n0 + j * 16 + lm;
                if constexpr (F16OUT)
                    ((_Float16*)Cout)[(size_t)row * N + col] = (_Float16)acc[i][j][r];
                else
                    ((float*)Cout)[(size_t)row * N + col] = acc[i][j][r];
            }
}

// ---------------------------------------------------------------------------
// NeoX RoPE + layout split:  qkv[b*s][6144] ->
//   Q[b][h][s][128] (rope), K[b][kv][s][128] (rope), V[b][kv][s][128] (copy)
// grid = (B*S, H + 2*KV), block = 64 (one thread per rotation pair)
// ---------------------------------------------------------------------------
__global__ __launch_bounds__(64) void rope_split_kernel(
    const _Float16* __restrict__ qkv, const int* __restrict__ pos,
    _Float16* __restrict__ Q, _Float16* __restrict__ K,
    _Float16* __restrict__ V) {
    const int S = 2048, D = 128, H = 32, KV = 8, W = 6144;
    const int bs = blockIdx.x;
    const int b = bs >> 11, s = bs & 2047;
    const int hh = blockIdx.y;
    const int i = threadIdx.x;                  // 0..63 (= D/2)
    const _Float16* row = qkv + (size_t)bs * W;
    const float p = (float)pos[bs];
    const float inv = __powf(10000.f, -(float)i * (1.f / 64.f));
    const float ang = p * inv;
    const float c = __cosf(ang), sn = __sinf(ang);

    if (hh < H) {
        const _Float16* src = row + hh * D;
        float x1 = (float)src[i], x2 = (float)src[i + 64];
        _Float16* dst = Q + (((size_t)b * H + hh) * S + s) * D;
        dst[i]      = (_Float16)(x1 * c - x2 * sn);
        dst[i + 64] = (_Float16)(x2 * c + x1 * sn);
    } else if (hh < H + KV) {
        int kh = hh - H;
        const _Float16* src = row + H * D + kh * D;
        float x1 = (float)src[i], x2 = (float)src[i + 64];
        _Float16* dst = K + (((size_t)b * KV + kh) * S + s) * D;
        dst[i]      = (_Float16)(x1 * c - x2 * sn);
        dst[i + 64] = (_Float16)(x2 * c + x1 * sn);
    } else {
        int vh = hh - H - KV;
        const _Float16* src = row + (H + KV) * D + vh * D;
        _Float16* dst = V + (((size_t)b * KV + vh) * S + s) * D;
        dst[i]      = src[i];
        dst[i + 64] = src[i + 64];
    }
}

// ---------------------------------------------------------------------------
// Flash-style causal GQA attention, all-WMMA.
//   grid = (S/128, H, B), block = 256 (8 waves, 16 q-rows per wave)
// QK^T: K's B-operand fragment is row-contiguous in [s][d] memory -> direct
// global b128 loads, no staging. PV: V transposed through LDS; P converted
// C-layout -> A-layout through per-wave LDS. Online softmax via half-wave
// shfl_xor reductions (C rows live across 16-lane groups on wave32).
// ---------------------------------------------------------------------------
__global__ __launch_bounds__(256) void attn_wmma_kernel(
    const _Float16* __restrict__ Q, const _Float16* __restrict__ K,
    const _Float16* __restrict__ V, _Float16* __restrict__ Aout) {
    const int S = 2048, D = 128, H = 32, KV = 8;
    const float SCALE = 0.08838834764831845f;   // 128^-0.5

    __shared__ _Float16 VT[128 * 32];           // [d][kv] transposed V tile
    __shared__ _Float16 Pls[8][16 * 32];        // per-wave P staging

    const int tid = threadIdx.x, lane = tid & 31, wave = tid >> 5;
    const int lm = lane & 15, hi = lane >> 4;
    const int qb = blockIdx.x * 128;
    const int h = blockIdx.y, b = blockIdx.z;
    const int kvh = h >> 2;                     // H/KV = 4

    const _Float16* Qp = Q + ((size_t)(b * H + h) * S) * D;
    const _Float16* Kp = K + ((size_t)(b * KV + kvh) * S) * D;
    const _Float16* Vp = V + ((size_t)(b * KV + kvh) * S) * D;
    const int qrow0 = qb + wave * 16;

    // Preload this wave's Q fragments (16 rows x 128 d = 4 k-steps)
    Frag qf[4];
    {
        const _Float16* qr = Qp + (size_t)(qrow0 + lm) * D;
#pragma unroll
        for (int t = 0; t < 4; ++t) {
            qf[t].q[0] = *(const int4*)(qr + t * 32 + hi * 8);
            qf[t].q[1] = *(const int4*)(qr + t * 32 + 16 + hi * 8);
        }
    }

    float mst[8], lst[8];
    v8f acc[8];
#pragma unroll
    for (int r = 0; r < 8; ++r) { mst[r] = -1e30f; lst[r] = 0.f; }
#pragma unroll
    for (int nt = 0; nt < 8; ++nt) acc[nt] = vzero8();

    const int kend = qb + 128;                  // causal upper bound for block
    for (int kv0 = 0; kv0 < kend; kv0 += 32) {
        __syncthreads();                        // prior VT reads done
        {   // stage V tile 32x128 transposed -> VT[d][kv]
            int r = tid >> 3, c = (tid & 7) * 16;
            const int4* src = (const int4*)(Vp + (size_t)(kv0 + r) * D + c);
            union { int4 q[2]; _Float16 h[16]; } tmp;
            tmp.q[0] = src[0]; tmp.q[1] = src[1];
#pragma unroll
            for (int j = 0; j < 16; ++j) VT[(c + j) * 32 + r] = tmp.h[j];
        }
        __syncthreads();

        // --- scores S = Q @ K^T (16 x 32), f32 ---
        v8f s0 = vzero8(), s1 = vzero8();
#pragma unroll
        for (int t = 0; t < 4; ++t) {
            Frag b0, b1;
            const _Float16* k0p = Kp + (size_t)(kv0 + lm) * D + t * 32 + hi * 8;
            b0.q[0] = ((const int4*)k0p)[0];
            b0.q[1] = ((const int4*)k0p)[2];
            const _Float16* k1p = Kp + (size_t)(kv0 + 16 + lm) * D + t * 32 + hi * 8;
            b1.q[0] = ((const int4*)k1p)[0];
            b1.q[1] = ((const int4*)k1p)[2];
            s0 = wmma_f16(qf[t].v, b0.v, s0);
            s1 = wmma_f16(qf[t].v, b1.v, s1);
        }

        // --- causal mask + online softmax (per C-layout row) ---
#pragma unroll
        for (int r = 0; r < 8; ++r) {
            const int qi = qrow0 + r + 8 * hi;
            float v0 = s0[r] * SCALE;
            float v1 = s1[r] * SCALE;
            if (kv0 + lm > qi)      v0 = -1e30f;
            if (kv0 + 16 + lm > qi) v1 = -1e30f;
            float rm = fmaxf(v0, v1);
#pragma unroll
            for (int msk = 1; msk < 16; msk <<= 1)
                rm = fmaxf(rm, __shfl_xor(rm, msk, 16));
            const float nm = fmaxf(mst[r], rm);
            const float alpha = __expf(mst[r] - nm);
            mst[r] = nm;
            const float p0 = __expf(v0 - nm);
            const float p1 = __expf(v1 - nm);
            float rs = p0 + p1;
#pragma unroll
            for (int msk = 1; msk < 16; msk <<= 1)
                rs += __shfl_xor(rs, msk, 16);
            lst[r] = lst[r] * alpha + rs;
#pragma unroll
            for (int nt = 0; nt < 8; ++nt) acc[nt][r] *= alpha;
            _Float16* pp = &Pls[wave][(r + 8 * hi) * 32 + lm];
            pp[0]  = (_Float16)p0;
            pp[16] = (_Float16)p1;
        }

        // --- O += P @ V (A-frag of P from LDS, B-frags from VT) ---
        Frag pf;
        {
            const _Float16* pb = &Pls[wave][lm * 32];
            pf.q[0] = *(const int4*)(pb + hi * 8);
            pf.q[1] = *(const int4*)(pb + 16 + hi * 8);
        }
#pragma unroll
        for (int nt = 0; nt < 8; ++nt) {
            Frag vf;
            const _Float16* vp = &VT[(nt * 16 + lm) * 32 + hi * 8];
            vf.q[0] = ((const int4*)vp)[0];
            vf.q[1] = ((const int4*)vp)[2];
            acc[nt] = wmma_f16(pf.v, vf.v, acc[nt]);
        }
    }

    // --- normalize and store as attn[b][s][h][d] f16 (input of O-proj) ---
#pragma unroll
    for (int nt = 0; nt < 8; ++nt)
#pragma unroll
        for (int r = 0; r < 8; ++r) {
            const int s = qrow0 + r + 8 * hi;
            const int d = nt * 16 + lm;
            const float o = acc[nt][r] / lst[r];
            Aout[(((size_t)b * S + s) * H + h) * D + d] = (_Float16)o;
        }
}

// ---------------------------------------------------------------------------
// Launch
// ---------------------------------------------------------------------------
extern "C" void kernel_launch(void* const* d_in, const int* in_sizes, int n_in,
                              void* d_out, int out_size, void* d_ws, size_t ws_size,
                              hipStream_t stream) {
    (void)in_sizes; (void)n_in; (void)out_size; (void)ws_size;
    const float* hs   = (const float*)d_in[0];   // (2,2048,4096) f32
    const int*   pos  = (const int*)d_in[1];     // (2,2048) i32
    const float* wqkv = (const float*)d_in[2];   // (4096,6144) f32
    const float* wo   = (const float*)d_in[3];   // (4096,4096) f32
    float* out = (float*)d_out;                  // (2,2048,4096) f32

    char* ws = (char*)d_ws;
    // Region A (33.5MB): hidden16, later reused as attn16
    _Float16* hidden16 = (_Float16*)(ws);
    _Float16* attn16   = hidden16;
    // Region B (50.3MB): wqkvT16, later reused as Q16 | K16 | V16
    _Float16* wqkvT16 = (_Float16*)(ws + 33554432ull);
    _Float16* Q16     = wqkvT16;
    _Float16* K16     = (_Float16*)(ws + 33554432ull + 33554432ull);
    _Float16* V16     = (_Float16*)(ws + 33554432ull + 41943040ull);
    // Region C (33.5MB): woT16
    _Float16* woT16 = (_Float16*)(ws + 83886080ull);
    // Region D (50.3MB): qkv16
    _Float16* qkv16 = (_Float16*)(ws + 117440512ull);

    // 1) activations: f32 -> f16; weights: f32 -> f16 + transpose (one-time)
    cvt_f32_f16_x4<<<2048, 256, 0, stream>>>((const float4*)hs, (v4h*)hidden16,
                                             16777216L / 4);
    cvt_transpose_f16<<<dim3(96, 64), 256, 0, stream>>>(wqkv, wqkvT16, 4096, 6144);
    cvt_transpose_f16<<<dim3(64, 64), 256, 0, stream>>>(wo, woT16, 4096, 4096);

    // 2) QKV projection: (4096x4096) @ (4096x6144) -> qkv16 f16
    gemm_f16_wmma<true, true><<<dim3(48, 32), 256, 0, stream>>>(
        hidden16, wqkvT16, (void*)qkv16, 4096, 6144, 4096);

    // 3) RoPE + split into Q/K/V (overwrites dead wqkvT16 region)
    rope_split_kernel<<<dim3(4096, 48), 64, 0, stream>>>(qkv16, pos, Q16, K16, V16);

    // 4) causal GQA flash attention -> attn16 (overwrites dead hidden16)
    attn_wmma_kernel<<<dim3(16, 32, 2), 256, 0, stream>>>(Q16, K16, V16, attn16);

    // 5) O projection: (4096x4096) @ (4096x4096) -> out f32
    gemm_f16_wmma<false, true><<<dim3(32, 32), 256, 0, stream>>>(
        attn16, woT16, (void*)out, 4096, 4096, 4096);
}
